// DA_conv_11115375362117
// MI455X (gfx1250) — matrix operand order repaired and checked by reading, not compiled, not run on hardware
//
#include <hip/hip_runtime.h>
#include <hip/hip_bf16.h>

typedef __attribute__((ext_vector_type(16))) __bf16 v16bf;
typedef __attribute__((ext_vector_type(8)))  __bf16 v8bf;
typedef __attribute__((ext_vector_type(8)))  float  v8f;

#define BB 16
#define CC 128
#define HH 128
#define WW 128
#define LD 136   // padded LDS/weight row stride in bf16 elements (multiple of 8 -> 16B aligned rows)

__device__ __forceinline__ float lrelu(float v) { return v > 0.f ? v : 0.1f * v; }

// ---------------------------------------------------------------------------
// Kernel A: tiny MLPs.
//  kern[b, c*9+j] = (lrelu(deg @ w1^T) @ w2^T)         (per-sample depthwise 3x3 kernels)
//  att[b, o]     = sigmoid(lrelu(deg @ ca_w1^T) @ ca_w2^T)
//  wbf           = conv_w converted to bf16, padded row stride LD (A-matrix source)
// ---------------------------------------------------------------------------
__global__ __launch_bounds__(128)
void prep_kernel(const float* __restrict__ deg, const float* __restrict__ w1,
                 const float* __restrict__ w2, const float* __restrict__ ca_w1,
                 const float* __restrict__ ca_w2, const float* __restrict__ conv_w,
                 float* __restrict__ kern, float* __restrict__ att,
                 __bf16* __restrict__ wbf)
{
    const int b = blockIdx.x;
    const int t = threadIdx.x;   // 0..127

    __shared__ float degs[CC];
    __shared__ float hid[CC];
    __shared__ float cah[16];

    degs[t] = deg[b * CC + t];
    __syncthreads();

    // hidden = lrelu(deg @ w1^T)
    float s = 0.f;
    for (int c = 0; c < CC; ++c) s += degs[c] * w1[t * CC + c];
    hid[t] = lrelu(s);

    if (t < 16) {
        float s2 = 0.f;
        for (int c = 0; c < CC; ++c) s2 += degs[c] * ca_w1[t * CC + c];
        cah[t] = lrelu(s2);
    }
    __syncthreads();

    // kern row: 1152 outputs, thread t computes o = t + 128*j
    for (int j = 0; j < 9; ++j) {
        const int o = t + CC * j;
        float s3 = 0.f;
        for (int k = 0; k < CC; ++k) s3 += hid[k] * w2[o * CC + k];
        kern[b * CC * 9 + o] = s3;
    }

    // channel attention
    float s4 = 0.f;
    for (int r = 0; r < 16; ++r) s4 += cah[r] * ca_w2[t * 16 + r];
    att[b * CC + t] = 1.f / (1.f + __expf(-s4));

    // conv_w (o,c) -> bf16 with padded stride; block b converts rows 8b..8b+7
    for (int q = 0; q < 8; ++q) {
        const int o = b * 8 + q;
        wbf[o * LD + t] = (__bf16)conv_w[o * CC + t];
    }
}

// ---------------------------------------------------------------------------
// Kernel B: fused per-row pipeline. One workgroup = one (b, h) row.
//   1) depthwise 3x3 (per-(b,c) kernel) + lrelu -> bf16 LDS tile t[w][c]
//   2) 1x1 conv as 128x128 x 128x128 bf16 WMMA (f32 accumulate)
//   3) epilogue: + conv_b + x0 * att, store f32
// ---------------------------------------------------------------------------
__global__ __launch_bounds__(256)
void fused_kernel(const float* __restrict__ x0, const float* __restrict__ conv_b,
                  const float* __restrict__ kern, const float* __restrict__ att,
                  const __bf16* __restrict__ wbf, float* __restrict__ out)
{
    __shared__ __align__(16) __bf16 t_lds[WW * LD];   // [w][c], padded
    __shared__ float kloc[CC * 9];

    const int h   = blockIdx.x;
    const int b   = blockIdx.y;
    const int tid = threadIdx.x;

    // stage per-channel 3x3 kernels for this batch into LDS
    for (int idx = tid; idx < CC * 9; idx += 256) kloc[idx] = kern[b * CC * 9 + idx];
    __syncthreads();

    // ---- depthwise 3x3 + lrelu -> bf16 LDS ----
    const int w0 = tid & (WW - 1);
    const int cb = tid >> 7;               // 0 or 1
    #pragma unroll 2
    for (int i = 0; i < 64; ++i) {
        const int c = cb + 2 * i;
        const float* kk   = &kloc[c * 9];
        const float* base = x0 + (((size_t)b * CC + c) * HH + h) * WW;
        float acc = 0.f;
        #pragma unroll
        for (int dy = -1; dy <= 1; ++dy) {
            const int hh = h + dy;
            if (hh < 0 || hh >= HH) continue;
            const float* row = base + (ptrdiff_t)dy * WW;
            const float* kr  = kk + (dy + 1) * 3;
            if (w0 > 0)      acc += kr[0] * row[w0 - 1];
            acc += kr[1] * row[w0];
            if (w0 < WW - 1) acc += kr[2] * row[w0 + 1];
        }
        t_lds[w0 * LD + c] = (__bf16)lrelu(acc);
    }
    __syncthreads();

    // ---- 1x1 conv: D(128xo,128xw) = W(128x128) x T(128x128), bf16 WMMA ----
    const int wave = tid >> 5;
    const int lane = tid & 31;
    const int lrow = lane & 15;
    const int half = lane >> 4;
    const int m0   = wave * 16;            // this wave's output-channel strip

    // A fragments: 16x32 bf16 tile of conv weights per K step.
    // ISA layout: element j -> K = k0 + 8*half + j (j<8), K = k0 + 16 + 8*half + (j-8) (j>=8)
    v16bf afrag[4];
    const __bf16* arow = wbf + (m0 + lrow) * LD;
    #pragma unroll
    for (int kt = 0; kt < 4; ++kt) {
        const int k0 = kt * 32;
        v8bf lo = *(const v8bf*)(arow + k0 + half * 8);
        v8bf hi = *(const v8bf*)(arow + k0 + 16 + half * 8);
        v16bf a;
        *(v8bf*)&a       = lo;
        *((v8bf*)&a + 1) = hi;
        afrag[kt] = a;
    }

    // epilogue constants for this wave's rows (C/D layout: M = m0 + r + 8*half)
    float bias_o[8], att_o[8];
    #pragma unroll
    for (int r = 0; r < 8; ++r) {
        const int o = m0 + r + 8 * half;
        bias_o[r] = conv_b[o];
        att_o[r]  = att[b * CC + o];
    }

    #pragma unroll
    for (int nt = 0; nt < 8; ++nt) {
        const int n0 = nt * 16;
        // B layout: element j -> K = k0 + 8*half + j (j<8), K = k0 + 16 + 8*half + (j-8) (j>=8)
        const __bf16* brow = &t_lds[(n0 + lrow) * LD + half * 8];
        v8f acc = {};
        #pragma unroll
        for (int kt = 0; kt < 4; ++kt) {
            v8bf lo = *(const v8bf*)(brow + kt * 32);
            v8bf hi = *(const v8bf*)(brow + kt * 32 + 16);
            v16bf bfr;
            *(v8bf*)&bfr       = lo;
            *((v8bf*)&bfr + 1) = hi;
            acc = __builtin_amdgcn_wmma_f32_16x16x32_bf16(
                /*neg_a=*/false, afrag[kt], /*neg_b=*/false, bfr,
                /*c_mod=*/(short)0, acc, /*reuse_a=*/false, /*reuse_b=*/false);
        }
        // fused epilogue: + bias + x0 * att
        #pragma unroll
        for (int r = 0; r < 8; ++r) {
            const int o = m0 + r + 8 * half;
            const int w = n0 + lrow;
            const size_t gi = (((size_t)b * CC + o) * HH + h) * WW + w;
            out[gi] = acc[r] + bias_o[r] + x0[gi] * att_o[r];
        }
    }
}

// ---------------------------------------------------------------------------
extern "C" void kernel_launch(void* const* d_in, const int* in_sizes, int n_in,
                              void* d_out, int out_size, void* d_ws, size_t ws_size,
                              hipStream_t stream) {
    const float* x0     = (const float*)d_in[0];
    const float* deg    = (const float*)d_in[1];
    const float* w1     = (const float*)d_in[2];
    const float* w2     = (const float*)d_in[3];
    const float* conv_w = (const float*)d_in[4];
    const float* conv_b = (const float*)d_in[5];
    const float* ca_w1  = (const float*)d_in[6];
    const float* ca_w2  = (const float*)d_in[7];
    float* out = (float*)d_out;

    // workspace layout
    float*  kern = (float*)d_ws;               // 16*1152 f32
    float*  att  = kern + BB * CC * 9;         // 16*128  f32
    __bf16* wbf  = (__bf16*)(att + BB * CC);   // 128*LD  bf16 (16B-aligned offset)

    prep_kernel<<<dim3(BB), dim3(128), 0, stream>>>(deg, w1, w2, ca_w1, ca_w2,
                                                    conv_w, kern, att, wbf);
    fused_kernel<<<dim3(HH, BB), dim3(256), 0, stream>>>(x0, conv_b, kern, att, wbf, out);
}
